// mol_atomstic_encoder_10058813407597
// MI455X (gfx1250) — matrix operand order repaired and checked by reading, not compiled, not run on hardware
//
#include <hip/hip_runtime.h>
#include <hip/hip_bf16.h>

// ---------------- problem dims (match reference) ----------------
#define NN      32768      // total nodes
#define NPG     64
#define BG      512        // graphs
#define DIM     128
#define HEADS   4
#define DHEAD   32
#define LAYERS  10
#define NE      131072     // edges
#define ATOMF   9
#define ATOMV   119
#define ATOMD   98
#define BONDF   3
#define BONDV   6
#define SASAH   10
#define PEO     20

typedef __attribute__((ext_vector_type(16))) __bf16          v16bf;
typedef __attribute__((ext_vector_type(16))) unsigned short  v16u;
typedef __attribute__((ext_vector_type(8)))  float           v8f;

union BFC  { v16u u; v16bf b; };
union Frag { uint4 q[2]; v16bf b; };    // two 16B loads -> one bf16 fragment

__device__ __forceinline__ unsigned short f2bf(float f) {
  unsigned u = __float_as_uint(f);
  u += 0x7FFFu + ((u >> 16) & 1u);      // round-to-nearest-even
  return (unsigned short)(u >> 16);
}

// ---- CDNA5 async global->LDS staging (ISA 08_async_tensor) ----
__device__ __forceinline__ void async_ld_b128(const void* gptr, void* lptr) {
  unsigned lds = (unsigned)(unsigned long long)lptr;      // low 32 bits = LDS byte offset
  unsigned long long ga = (unsigned long long)gptr;
  asm volatile("global_load_async_to_lds_b128 %0, %1, off" :: "v"(lds), "v"(ga) : "memory");
}
__device__ __forceinline__ void wait_async0() {
  asm volatile("s_wait_asynccnt 0" ::: "memory");
}

// ---- f32-source fragment packers (attention LDS tiles) ----
// A fragment (16x32): lane half, VGPR v<4 -> K = k0+8*half+2v ; v>=4 -> K = k0+16+8*half+2(v-4)
__device__ __forceinline__ v16bf pack_a_rowk(const float* rowbase, int k0, int half) {
  v16u r;
#pragma unroll
  for (int v = 0; v < 8; ++v) {
    int k = k0 + ((v < 4) ? (8 * half + 2 * v) : (16 + 8 * half + 2 * (v - 4)));
    float2 p = *(const float2*)(rowbase + k);
    r[2 * v]     = f2bf(p.x);
    r[2 * v + 1] = f2bf(p.y);
  }
  BFC c; c.u = r; return c.b;
}
// B fragment (32x16): lane half, VGPR v<4 -> K = k0+8*half+2v ; v>=4 -> K = k0+16+8*half+2(v-4). Column n contiguous over K at colbase.
__device__ __forceinline__ v16bf pack_b_rowk(const float* colbase, int k0, int half) {
  v16u r;
#pragma unroll
  for (int v = 0; v < 8; ++v) {
    int k = k0 + ((v < 4) ? (8 * half + 2 * v) : (16 + 8 * half + 2 * (v - 4)));
    float2 p = *(const float2*)(colbase + k);
    r[2 * v]     = f2bf(p.x);
    r[2 * v + 1] = f2bf(p.y);
  }
  BFC c; c.u = r; return c.b;
}
// B fragment where B[k][n] = src[k*stride + n] (K strided)
__device__ __forceinline__ v16bf pack_b_strided(const float* src, int stride, int n, int k0, int half) {
  v16u r;
#pragma unroll
  for (int v = 0; v < 8; ++v) {
    int k = k0 + ((v < 4) ? (8 * half + 2 * v) : (16 + 8 * half + 2 * (v - 4)));
    r[2 * v]     = f2bf(src[(size_t)k * stride + n]);
    r[2 * v + 1] = f2bf(src[(size_t)(k + 1) * stride + n]);
  }
  BFC c; c.u = r; return c.b;
}

// ---------------- f32 -> bf16 streaming conversion (x4 per thread) ----------------
__global__ __launch_bounds__(256) void cvt_bf16x4(const float* __restrict__ src,
                                                  unsigned short* __restrict__ dst, int n4) {
  int i = blockIdx.x * 256 + threadIdx.x;
  if (i >= n4) return;
  float4 f = ((const float4*)src)[i];
  ushort4 u;
  u.x = f2bf(f.x); u.y = f2bf(f.y); u.z = f2bf(f.z); u.w = f2bf(f.w);
  ((ushort4*)dst)[i] = u;
}

// ---------------- feature fusion ----------------
__global__ __launch_bounds__(256) void fuse_features(
    const int* __restrict__ x, const float* __restrict__ sasa, const float* __restrict__ pe,
    const float* __restrict__ atom_emb, const float* __restrict__ sasa_w, const float* __restrict__ sasa_b,
    const float* __restrict__ pe_w, const float* __restrict__ pe_b, float* __restrict__ h) {
  int idx = blockIdx.x * 256 + threadIdx.x;            // over NN*DIM
  int n = idx >> 7, d = idx & 127;
  float acc;
  if (d < ATOMD) {
    acc = 0.f;
#pragma unroll
    for (int f = 0; f < ATOMF; ++f) {
      int tok = x[n * ATOMF + f];
      acc += atom_emb[((size_t)f * ATOMV + tok) * ATOMD + d];
    }
  } else if (d < ATOMD + SASAH) {
    int j = d - ATOMD;
    acc = sasa_b[j];
#pragma unroll
    for (int c = 0; c < 4; ++c) {
      float s = sasa[n * 4 + c];
      if (s != s) s = 0.f;                              // nan_to_num
      s = fminf(fmaxf(s, -3.402823466e38f), 3.402823466e38f);
      acc += s * sasa_w[j * 4 + c];
    }
  } else {
    int j = d - ATOMD - SASAH;
    acc = pe_b[j];
#pragma unroll
    for (int c = 0; c < PEO; ++c)
      acc += pe[n * PEO + c] * pe_w[j * PEO + c];
  }
  h[idx] = acc;
}

__global__ __launch_bounds__(256) void edge_embed(
    const int* __restrict__ attr, const float* __restrict__ bond_emb, float* __restrict__ e) {
  int idx = blockIdx.x * 256 + threadIdx.x;            // over NE*DIM
  int ed = idx >> 7, d = idx & 127;
  float acc = 0.f;
#pragma unroll
  for (int f = 0; f < BONDF; ++f)
    acc += bond_emb[((size_t)f * BONDV + attr[ed * BONDF + f]) * DIM + d];
  e[idx] = acc;
}

// ---------------- elementwise + GINE scatter ----------------
__global__ __launch_bounds__(256) void ew_copy(float* __restrict__ dst, const float* __restrict__ src, int n) {
  int i = blockIdx.x * 256 + threadIdx.x;
  if (i < n) dst[i] = src[i];
}
__global__ __launch_bounds__(256) void ew_add(float* __restrict__ dst, const float* __restrict__ a,
                                              const float* __restrict__ b, int n) {
  int i = blockIdx.x * 256 + threadIdx.x;
  if (i < n) dst[i] = a[i] + b[i];
}
__global__ __launch_bounds__(256) void gine_scatter(
    const float* __restrict__ h, const float* __restrict__ e,
    const int* __restrict__ src, const int* __restrict__ dst, float* __restrict__ z) {
  int idx = blockIdx.x * 256 + threadIdx.x;            // over NE*DIM
  int ed = idx >> 7, d = idx & 127;
  float m = h[(size_t)src[ed] * DIM + d] + e[idx];
  m = fmaxf(m, 0.f);
  atomicAdd(&z[(size_t)dst[ed] * DIM + d], m);
}

// ---------------- WMMA GEMM on pre-converted bf16 ----------------
// C[M,Nout] = act(A[M,K] @ W[Nout,K]^T + bias); A,W row-major bf16.
// Fragment fetch is pure b128 loads: A frag = 16B @ (k0+8*half) and 16B @ (k0+16+8*half);
// B frag = 16B @ (k0+8*half) and 16B @ (k0+16+8*half).  No per-element conversion.
__global__ __launch_bounds__(256) void gemm_bf16(
    const unsigned short* __restrict__ A, const unsigned short* __restrict__ W,
    const float* __restrict__ bias, float* __restrict__ C, int M, int K, int Nout, int relu) {
  int wave = threadIdx.x >> 5;
  int lane = threadIdx.x & 31;
  int lm = lane & 15, half = lane >> 4;
  int tile   = blockIdx.x * 8 + wave;
  int tilesN = Nout >> 4;
  int tm = tile / tilesN, tn = tile % tilesN;
  const unsigned short* arow = A + (size_t)(tm * 16 + lm) * K;
  const unsigned short* wrow = W + (size_t)(tn * 16 + lm) * K;
  v8f acc = {};
  for (int k0 = 0; k0 < K; k0 += 32) {
    Frag fa, fb;
    fa.q[0] = *(const uint4*)(arow + k0 + 8 * half);
    fa.q[1] = *(const uint4*)(arow + k0 + 16 + 8 * half);
    fb.q[0] = *(const uint4*)(wrow + k0 + 8 * half);
    fb.q[1] = *(const uint4*)(wrow + k0 + 16 + 8 * half);
    acc = __builtin_amdgcn_wmma_f32_16x16x32_bf16(false, fa.b, false, fb.b, (short)0, acc, false, false);
  }
  int c = tn * 16 + lm;
  float b = bias[c];
#pragma unroll
  for (int v = 0; v < 8; ++v) {                        // C layout: row = v + 8*half, col = lm
    int r = tm * 16 + v + 8 * half;
    float val = acc[v] + b;
    if (relu) val = fmaxf(val, 0.f);
    C[(size_t)r * Nout + c] = val;
  }
}

// ---------------- attention: per (graph, head) block ----------------
__global__ __launch_bounds__(256) void attention_kernel(
    const float* __restrict__ qkv, const float* __restrict__ sph,
    const float* __restrict__ hop, float* __restrict__ ao) {
  __shared__ float sQ[NPG][DHEAD];
  __shared__ float sK[NPG][DHEAD];
  __shared__ float sV[NPG][DHEAD];
  __shared__ float sS[NPG][NPG];
  int b  = blockIdx.x / HEADS;
  int hd = blockIdx.x % HEADS;
  int tid = threadIdx.x;
  const float* base = qkv + (size_t)b * NPG * (3 * DIM) + hd * DHEAD;
  // async-stage Q/K/V tiles into LDS: 16B per lane per issue, rows are 128B contiguous
#pragma unroll
  for (int it = 0; it < 2; ++it) {
    int row = (tid >> 3) + it * 32;
    int seg = (tid & 7) * 4;                           // float offset within row (16B granules)
    const float* g = base + (size_t)row * (3 * DIM);
    async_ld_b128(g + seg,            &sQ[row][seg]);
    async_ld_b128(g + DIM + seg,      &sK[row][seg]);
    async_ld_b128(g + 2 * DIM + seg,  &sV[row][seg]);
  }
  wait_async0();
  __syncthreads();

  int wave = tid >> 5, lane = tid & 31, lm = lane & 15, half = lane >> 4;
  const float scale = 0.17677669529663687f;            // 1/sqrt(32)

  // S = (Q K^T) * scale : 4x4 tiles of 16x16, K=32; 2 tiles per wave
  for (int t = wave; t < 16; t += 8) {
    int ti = t >> 2, tj = t & 3;
    v16bf af = pack_a_rowk(&sQ[ti * 16 + lm][0], 0, half);
    v16bf bf = pack_b_rowk(&sK[tj * 16 + lm][0], 0, half); // B[k,n] = K[n,k]
    v8f acc = {};
    acc = __builtin_amdgcn_wmma_f32_16x16x32_bf16(false, af, false, bf, (short)0, acc, false, false);
#pragma unroll
    for (int v = 0; v < 8; ++v)
      sS[ti * 16 + v + 8 * half][tj * 16 + lm] = acc[v] * scale;
  }
  __syncthreads();

  // row softmax, then post-softmax hop decay: * 0.5^relu(sph - hop)
  if (tid < NPG) {
    int i = tid;
    float hv = hop[hd];
    const float* srow = sph + ((size_t)b * NPG + i) * NPG;
    float mx = -3.402823466e38f;
    for (int j = 0; j < NPG; ++j) mx = fmaxf(mx, sS[i][j]);
    float sum = 0.f;
    for (int j = 0; j < NPG; ++j) { float ex = __expf(sS[i][j] - mx); sS[i][j] = ex; sum += ex; }
    float inv = 1.f / sum;
    for (int j = 0; j < NPG; ++j) {
      float dd = fmaxf(srow[j] - hv, 0.f);             // leaky_relu slope 0 == relu
      sS[i][j] = sS[i][j] * inv * exp2f(-dd);          // gamma=0.5 -> 2^-d
    }
  }
  __syncthreads();

  // AO = attn @ V : 64x64 @ 64x32 -> 8 tiles (4x2), one per wave, K=64 (2 WMMAs)
  {
    int ti = wave >> 1, tj = wave & 1;
    v8f acc = {};
#pragma unroll
    for (int k0 = 0; k0 < 64; k0 += 32) {
      v16bf af = pack_a_rowk(&sS[ti * 16 + lm][0], k0, half);
      v16bf bf = pack_b_strided(&sV[0][0], DHEAD, tj * 16 + lm, k0, half); // B[k,n] = V[k,n]
      acc = __builtin_amdgcn_wmma_f32_16x16x32_bf16(false, af, false, bf, (short)0, acc, false, false);
    }
    float* aorow = ao + (size_t)b * NPG * DIM + hd * DHEAD;
#pragma unroll
    for (int v = 0; v < 8; ++v) {
      int r = ti * 16 + v + 8 * half;
      aorow[(size_t)r * DIM + tj * 16 + lm] = acc[v];
    }
  }
}

// ---------------- BatchNorm (training-mode, affine identity) ----------------
__global__ __launch_bounds__(256) void zero_stats(float* __restrict__ stats) {
  if (threadIdx.x < 2 * DIM) stats[threadIdx.x] = 0.f;
}
__global__ __launch_bounds__(256) void bn_stats(
    const float* __restrict__ a, const float* __restrict__ b, float* __restrict__ stats, int total) {
  __shared__ float ssum[DIM], ssq[DIM];
  int tid = threadIdx.x;
  if (tid < DIM) { ssum[tid] = 0.f; ssq[tid] = 0.f; }
  __syncthreads();
  for (int i = blockIdx.x * 256 + tid; i < total; i += gridDim.x * 256) {
    float v = a[i] + b[i];
    int c = i & (DIM - 1);
    atomicAdd(&ssum[c], v);
    atomicAdd(&ssq[c], v * v);
  }
  __syncthreads();
  if (tid < DIM) {
    atomicAdd(&stats[tid], ssum[tid]);
    atomicAdd(&stats[DIM + tid], ssq[tid]);
  }
}
__global__ __launch_bounds__(256) void bn_apply(
    float* __restrict__ y, const float* __restrict__ a, const float* __restrict__ b,
    const float* __restrict__ stats, int total) {
  int i = blockIdx.x * 256 + threadIdx.x;
  if (i >= total) return;
  int c = i & (DIM - 1);
  const float invN = 1.f / (float)NN;
  float mean = stats[c] * invN;
  float var  = stats[DIM + c] * invN - mean * mean;
  float v = a[i] + b[i];
  y[i] = (v - mean) * rsqrtf(var + 1e-5f);
}

__global__ __launch_bounds__(256) void write_out(const float* __restrict__ h, float* __restrict__ out) {
  int i = blockIdx.x * 256 + threadIdx.x;              // over NN*DIM + NN
  if (i < NN * DIM) out[i] = h[i];
  else if (i < NN * DIM + NN) out[i] = 1.0f;           // full mask -> 1.0
}

// ---------------- host orchestration ----------------
extern "C" void kernel_launch(void* const* d_in, const int* in_sizes, int n_in,
                              void* d_out, int out_size, void* d_ws, size_t ws_size,
                              hipStream_t stream) {
  (void)in_sizes; (void)n_in; (void)out_size; (void)ws_size;
  const int*   x        = (const int*)  d_in[0];
  const float* sasa     = (const float*)d_in[1];
  const float* pe       = (const float*)d_in[2];
  const int*   eidx     = (const int*)  d_in[3];
  const int*   eattr    = (const int*)  d_in[4];
  /* d_in[5] = batch (uniform graphs -> unused) */
  const float* sph      = (const float*)d_in[6];
  const float* atom_emb = (const float*)d_in[7];
  const float* bond_emb = (const float*)d_in[8];
  const float* sasa_w   = (const float*)d_in[9];
  const float* sasa_b   = (const float*)d_in[10];
  const float* pe_w     = (const float*)d_in[11];
  const float* pe_b     = (const float*)d_in[12];
  const float* hop      = (const float*)d_in[13];
  const float* lin1_w   = (const float*)d_in[14];
  const float* lin1_b   = (const float*)d_in[15];
  const float* lin2_w   = (const float*)d_in[16];
  const float* lin2_b   = (const float*)d_in[17];
  const float* qkv_w    = (const float*)d_in[18];
  const float* qkv_b    = (const float*)d_in[19];
  const float* out_w    = (const float*)d_in[20];
  const float* out_b    = (const float*)d_in[21];
  const float* mlp1_w   = (const float*)d_in[22];
  const float* mlp1_b   = (const float*)d_in[23];
  const float* mlp2_w   = (const float*)d_in[24];
  const float* mlp2_b   = (const float*)d_in[25];

  const int ND = NN * DIM;           // 4,194,304
  const int ED = NE * DIM;           // 16,777,216
  const int DD = DIM * DIM;          // 16,384
  float* W = (float*)d_ws;
  size_t o = 0;
  float* h     = W + o; o += ND;
  float* ebuf  = W + o; o += (size_t)ED;
  float* z     = W + o; o += ND;     // z / "out" combine buffer
  float* s1    = W + o; o += (size_t)2 * ND;  // lin1-out / attn-out / mlp1-out
  float* s2    = W + o; o += ND;     // pre-BN buffers
  float* hl    = W + o; o += ND;
  float* ha    = W + o; o += ND;
  float* qkvB  = W + o; o += (size_t)3 * ND;
  float* stats = W + o; o += 2 * DIM;
  unsigned short* abf = (unsigned short*)(W + o); o += ND;        // bf16 activations (2*ND halves)
  unsigned short* wbf = (unsigned short*)(W + o); o += 5 * DD;    // bf16 weights (10*DD halves)

  unsigned short* wl1 = wbf;              // DD
  unsigned short* wl2 = wbf + DD;         // DD
  unsigned short* wqk = wbf + 2 * DD;     // 3*DD
  unsigned short* wou = wbf + 5 * DD;     // DD
  unsigned short* wm1 = wbf + 6 * DD;     // 2*DD
  unsigned short* wm2 = wbf + 8 * DD;     // 2*DD

  const int TPB = 256;
  fuse_features<<<ND / TPB, TPB, 0, stream>>>(x, sasa, pe, atom_emb, sasa_w, sasa_b, pe_w, pe_b, h);
  edge_embed<<<ED / TPB, TPB, 0, stream>>>(eattr, bond_emb, ebuf);

  const int GB128 = (NN / 16) * (DIM / 16) / 8;        // 2048 blocks
  const int GB256 = (NN / 16) * (2 * DIM / 16) / 8;    // 4096
  const int GB384 = (NN / 16) * (3 * DIM / 16) / 8;    // 6144
  const int CV_ND = (ND / 4) / TPB;                    // cvt grid for ND elems
  const int CV_2ND = (2 * ND / 4) / TPB;

  for (int l = 0; l < LAYERS; ++l) {
    const float* l1w = lin1_w + (size_t)l * DD;
    const float* l1b = lin1_b + (size_t)l * DIM;
    const float* l2w = lin2_w + (size_t)l * DD;
    const float* l2b = lin2_b + (size_t)l * DIM;
    const float* qw  = qkv_w  + (size_t)l * 3 * DD;
    const float* qb  = qkv_b  + (size_t)l * 3 * DIM;
    const float* ow  = out_w  + (size_t)l * DD;
    const float* ob  = out_b  + (size_t)l * DIM;
    const float* m1w = mlp1_w + (size_t)l * 2 * DD;
    const float* m1b = mlp1_b + (size_t)l * 2 * DIM;
    const float* m2w = mlp2_w + (size_t)l * 2 * DD;
    const float* m2b = mlp2_b + (size_t)l * DIM;

    // per-layer weight conversion (tiny)
    cvt_bf16x4<<<(DD / 4 + TPB - 1) / TPB, TPB, 0, stream>>>(l1w, wl1, DD / 4);
    cvt_bf16x4<<<(DD / 4 + TPB - 1) / TPB, TPB, 0, stream>>>(l2w, wl2, DD / 4);
    cvt_bf16x4<<<(3 * DD / 4 + TPB - 1) / TPB, TPB, 0, stream>>>(qw, wqk, 3 * DD / 4);
    cvt_bf16x4<<<(DD / 4 + TPB - 1) / TPB, TPB, 0, stream>>>(ow, wou, DD / 4);
    cvt_bf16x4<<<(2 * DD / 4 + TPB - 1) / TPB, TPB, 0, stream>>>(m1w, wm1, 2 * DD / 4);
    cvt_bf16x4<<<(2 * DD / 4 + TPB - 1) / TPB, TPB, 0, stream>>>(m2w, wm2, 2 * DD / 4);

    // ---- local GINE: z = h + segment_sum(relu(h[src]+e), dst)
    ew_copy<<<ND / TPB, TPB, 0, stream>>>(z, h, ND);
    gine_scatter<<<ED / TPB, TPB, 0, stream>>>(h, ebuf, eidx, eidx + NE, z);
    // hl = BN( relu(z@lin1^T+b)@lin2^T+b + h )
    cvt_bf16x4<<<CV_ND, TPB, 0, stream>>>(z, abf, ND / 4);
    gemm_bf16<<<GB128, TPB, 0, stream>>>(abf, wl1, l1b, s1, NN, DIM, DIM, 1);
    cvt_bf16x4<<<CV_ND, TPB, 0, stream>>>(s1, abf, ND / 4);
    gemm_bf16<<<GB128, TPB, 0, stream>>>(abf, wl2, l2b, s2, NN, DIM, DIM, 0);
    zero_stats<<<1, TPB, 0, stream>>>(stats);
    bn_stats<<<1024, TPB, 0, stream>>>(s2, h, stats, ND);
    bn_apply<<<ND / TPB, TPB, 0, stream>>>(hl, s2, h, stats, ND);
    // ---- global attention with hop decay
    cvt_bf16x4<<<CV_ND, TPB, 0, stream>>>(h, abf, ND / 4);
    gemm_bf16<<<GB384, TPB, 0, stream>>>(abf, wqk, qb, qkvB, NN, DIM, 3 * DIM, 0);
    attention_kernel<<<BG * HEADS, TPB, 0, stream>>>(qkvB, sph, hop, s1);
    cvt_bf16x4<<<CV_ND, TPB, 0, stream>>>(s1, abf, ND / 4);
    gemm_bf16<<<GB128, TPB, 0, stream>>>(abf, wou, ob, s2, NN, DIM, DIM, 0);
    zero_stats<<<1, TPB, 0, stream>>>(stats);
    bn_stats<<<1024, TPB, 0, stream>>>(s2, h, stats, ND);
    bn_apply<<<ND / TPB, TPB, 0, stream>>>(ha, s2, h, stats, ND);
    // ---- combine + MLP + final BN
    ew_add<<<ND / TPB, TPB, 0, stream>>>(z, hl, ha, ND);
    cvt_bf16x4<<<CV_ND, TPB, 0, stream>>>(z, abf, ND / 4);
    gemm_bf16<<<GB256, TPB, 0, stream>>>(abf, wm1, m1b, s1, NN, DIM, 2 * DIM, 1);
    cvt_bf16x4<<<CV_2ND, TPB, 0, stream>>>(s1, abf, 2 * ND / 4);
    gemm_bf16<<<GB128, TPB, 0, stream>>>(abf, wm2, m2b, s2, NN, 2 * DIM, DIM, 0);
    zero_stats<<<1, TPB, 0, stream>>>(stats);
    bn_stats<<<1024, TPB, 0, stream>>>(s2, z, stats, ND);
    bn_apply<<<ND / TPB, TPB, 0, stream>>>(h, s2, z, stats, ND);
  }

  write_out<<<(ND + NN) / TPB, TPB, 0, stream>>>(h, (float*)d_out);
}
